// RNNEncoder_2869038154544
// MI455X (gfx1250) — matrix operand order repaired and checked by reading, not compiled, not run on hardware
//
#include <hip/hip_runtime.h>

// ---------------------------------------------------------------------------
// 2-layer tanh RNN, T=1024, B=64, E=H=512 on gfx1250 (wave32, WMMA).
//   preX[t] = emb[input[t]] @ Wih0^T + (bih0+bhh0)      -- parallel WMMA GEMM
//   h0'     = tanh(preX[t] + h0 @ Whh0^T)               -- persistent kernel
//   h1'     = tanh([h0',h1] @ [Wih1;Whh1]^T + b1)
// Weights bf16, accumulate f32. Recurrent weights resident in LDS.
// Single grid barrier per step: phase2(t) and phase1(t+1) share one barrier
// interval and share A-fragments of h0' (each A load feeds 4 WMMAs).
// ---------------------------------------------------------------------------

#define T_STEPS 1024
#define BATCH   64
#define VOCAB   32000
#define EDIM    512
#define HDIM    512
#define NWG     16          // workgroups in persistent kernel
#define NCOL    32          // H columns per WG
#define P1E     520         // padded row pitch (elems) for K=512 weights  (1040B % 256 == 16)
#define P2E     1032        // padded row pitch (elems) for K=1024 weights (2064B % 256 == 16)

typedef __attribute__((ext_vector_type(16))) __bf16 v16bf;
typedef __attribute__((ext_vector_type(8)))  __bf16 v8bf;
typedef __attribute__((ext_vector_type(8)))  float  v8f;

// A-fragment (16x32 bf16): lane holds K 0..15 (lanes 0-15, M=lane)
// or K 16..31 (lanes 16-31). Two 16B loads at +0 and +8 elems.
static __device__ __forceinline__ v16bf ld_frag_a(const __bf16* p) {
  v8bf lo = *(const v8bf*)p;
  v8bf hi = *(const v8bf*)(p + 8);
  return __builtin_shufflevector(lo, hi, 0,1,2,3,4,5,6,7,8,9,10,11,12,13,14,15);
}
// B-fragment (32x16 bf16): lane holds 16 contiguous K for fixed N (=row of W).
static __device__ __forceinline__ v16bf ld_frag_b(const __bf16* p) {
  v8bf lo = *(const v8bf*)p;
  v8bf hi = *(const v8bf*)(p + 8);
  return __builtin_shufflevector(lo, hi, 0,1,2,3,4,5,6,7,8,9,10,11,12,13,14,15);
}
static __device__ __forceinline__ v8f wmma_bf16(v16bf a, v16bf b, v8f c) {
  return __builtin_amdgcn_wmma_f32_16x16x32_bf16(false, a, false, b, (short)0, c, false, false);
}
// CDNA5 TRANS-unit tanh if the builtin exists; libm fallback otherwise.
static __device__ __forceinline__ float fast_tanh(float x) {
#if __has_builtin(__builtin_amdgcn_tanhf)
  return __builtin_amdgcn_tanhf(x);
#else
  return tanhf(x);
#endif
}

// Sense-reversing global barrier over NWG co-resident workgroups.
static __device__ __forceinline__ void grid_barrier(unsigned* cnt, unsigned* gen) {
  __syncthreads();
  if (threadIdx.x == 0) {
    __threadfence();
    unsigned g = __hip_atomic_load(gen, __ATOMIC_RELAXED, __HIP_MEMORY_SCOPE_AGENT);
    unsigned t = __hip_atomic_fetch_add(cnt, 1u, __ATOMIC_ACQ_REL, __HIP_MEMORY_SCOPE_AGENT);
    if (t == NWG - 1) {
      __hip_atomic_store(cnt, 0u, __ATOMIC_RELAXED, __HIP_MEMORY_SCOPE_AGENT);
      __hip_atomic_fetch_add(gen, 1u, __ATOMIC_RELEASE, __HIP_MEMORY_SCOPE_AGENT);
    } else {
      while (__hip_atomic_load(gen, __ATOMIC_ACQUIRE, __HIP_MEMORY_SCOPE_AGENT) == g)
        __builtin_amdgcn_s_sleep(1);
    }
  }
  __syncthreads();
}

// ---------------------------------------------------------------------------
// Kernel 1: one-time packing. emb->bf16, Wih0->bf16 row-major,
// Whh0 / [Wih1;Whh1] -> bf16 per-WG LDS-image slices (padded pitch),
// fused biases b0=bih0+bhh0, b1=bih1+bhh1.
// ---------------------------------------------------------------------------
__global__ void pack_kernel(const float* __restrict__ emb,  const float* __restrict__ Wih0,
                            const float* __restrict__ Whh0, const float* __restrict__ Wih1,
                            const float* __restrict__ Whh1,
                            const float* __restrict__ bih0, const float* __restrict__ bhh0,
                            const float* __restrict__ bih1, const float* __restrict__ bhh1,
                            __bf16* __restrict__ embbf,  __bf16* __restrict__ wih0bf,
                            __bf16* __restrict__ whh0pk, __bf16* __restrict__ w1pk,
                            float* __restrict__ b0, float* __restrict__ b1) {
  const long stride = (long)gridDim.x * blockDim.x;
  const long gid    = (long)blockIdx.x * blockDim.x + threadIdx.x;
  for (long i = gid; i < (long)VOCAB * EDIM; i += stride) embbf[i]  = (__bf16)emb[i];
  for (long i = gid; i < (long)HDIM  * EDIM; i += stride) wih0bf[i] = (__bf16)Wih0[i];
  for (long i = gid; i < (long)HDIM * HDIM; i += stride) {
    int n = (int)(i >> 9), k = (int)(i & 511);
    whh0pk[(long)(n >> 5) * (NCOL * P1E) + (long)(n & 31) * P1E + k] = (__bf16)Whh0[i];
  }
  for (long i = gid; i < (long)HDIM * 1024; i += stride) {
    int n = (int)(i >> 10), k = (int)(i & 1023);
    float v = (k < 512) ? Wih1[((long)n << 9) + k] : Whh1[((long)n << 9) + (k - 512)];
    w1pk[(long)(n >> 5) * (NCOL * P2E) + (long)(n & 31) * P2E + k] = (__bf16)v;
  }
  for (long i = gid; i < HDIM; i += stride) {
    b0[i] = bih0[i] + bhh0[i];
    b1[i] = bih1[i] + bhh1[i];
  }
}

// ---------------------------------------------------------------------------
// Kernel 2: preX = gather(emb,input) @ Wih0^T + b0, stored as bf16 C-fragments
// (tile = mt*32+nt, per-lane 8 contiguous bf16). Fully parallel WMMA GEMM,
// embedding gather fused into the A-fragment address.
// grid (4096, 4), block 256 (8 waves): wave -> one 16x16 tile, K=512.
// ---------------------------------------------------------------------------
__global__ void prex_kernel(const int* __restrict__ tok, const __bf16* __restrict__ embbf,
                            const __bf16* __restrict__ wih0bf, const float* __restrict__ b0,
                            __bf16* __restrict__ preX) {
  const int wave = threadIdx.x >> 5, lane = threadIdx.x & 31;
  const int lm = lane & 15, hi = lane >> 4;
  const int mt = blockIdx.x;
  const int nt = blockIdx.y * 8 + wave;
  const int row = mt * 16 + lm;                 // flat (t*B + b) row
  const int t = tok[row];
  const __bf16* arow = embbf + (long)t * EDIM;
  const int n = nt * 16 + lm;
  const __bf16* brow = wih0bf + (long)n * EDIM;
  const float bias = b0[n];
  v8f c;
#pragma unroll
  for (int r = 0; r < 8; ++r) c[r] = bias;
#pragma unroll 4
  for (int kt = 0; kt < 16; ++kt) {
    v16bf a = ld_frag_a(arow + kt * 32 + hi * 16);
    v16bf b = ld_frag_b(brow + kt * 32 + hi * 16);
    c = wmma_bf16(a, b, c);
  }
  __bf16* dst = preX + ((long)(mt * 32 + nt) * 256 + lane * 8);
#pragma unroll
  for (int r = 0; r < 8; ++r) dst[r] = (__bf16)c[r];
}

// ---------------------------------------------------------------------------
// Kernel 3: persistent recurrence. 16 WGs x 128 threads (4 waves).
// Each WG owns 32 H-columns; weight slices live in LDS (97KB, padded rows ->
// conflict-free ds_load_b128). h state double-buffered in global (L2).
// One barrier per step; phase2(t) and phase1(t+1) merged.
// ---------------------------------------------------------------------------
__global__ void __launch_bounds__(128, 1)
rnn_kernel(const __bf16* __restrict__ preX,
           const __bf16* __restrict__ whh0pk, const __bf16* __restrict__ w1pk,
           const float* __restrict__ b1g,
           __bf16* h0a, __bf16* h0b, __bf16* h1a, __bf16* h1b,
           unsigned* bar, float* __restrict__ out) {
  __shared__ __align__(16) __bf16 smem[NCOL * P1E + NCOL * P2E];   // 99,328 bytes
  const int wg = blockIdx.x;
  const int tid = threadIdx.x;
  const int wave = tid >> 5, lane = tid & 31, lm = lane & 15, hi = lane >> 4;

  { // one-time LDS weight preload
    const uint4* s0 = (const uint4*)(whh0pk + (long)wg * (NCOL * P1E));
    uint4* d0 = (uint4*)smem;
    for (int i = tid; i < (NCOL * P1E * 2) / 16; i += 128) d0[i] = s0[i];
    const uint4* s1 = (const uint4*)(w1pk + (long)wg * (NCOL * P2E));
    uint4* d1 = (uint4*)(smem + NCOL * P1E);
    for (int i = tid; i < (NCOL * P2E * 2) / 16; i += 128) d1[i] = s1[i];
  }
  __syncthreads();

  const float bias1_0 = b1g[wg * NCOL + lm];
  const float bias1_1 = b1g[wg * NCOL + 16 + lm];
  __bf16* h0buf[2] = { h0a, h0b };
  __bf16* h1buf[2] = { h1a, h1b };
  const __bf16* ldsW0 = smem;                  // [32][P1E]
  const __bf16* ldsW1 = smem + NCOL * P1E;     // [32][P2E]
  unsigned* cnt = bar;
  unsigned* gen = bar + 64;                    // separate 256B line

  const long rowoff = (long)(wave * 16 + lm) * HDIM;
  const int  rbase  = wave * 16 + hi * 8;
  const int  col0   = wg * NCOL + lm;

  // ---- prologue: phase1(0): h0' = tanh(preX[0] + 0 @ Whh0^T) -> h0buf[1]
  {
    v8f d0, d1;
    const __bf16* f = preX + (((long)(wave) * 32 + wg * 2) * 256 + lane * 8);
    v8bf p0 = *(const v8bf*)f;
    v8bf p1 = *(const v8bf*)(f + 256);
#pragma unroll
    for (int r = 0; r < 8; ++r) { d0[r] = (float)p0[r]; d1[r] = (float)p1[r]; }
    const __bf16* hA = h0buf[0] + rowoff;      // zeros
#pragma unroll 4
    for (int kt = 0; kt < 16; ++kt) {
      v16bf a   = ld_frag_a(hA + kt * 32 + hi * 16);
      v16bf bf0 = ld_frag_b(ldsW0 + lm * P1E        + kt * 32 + hi * 16);
      v16bf bf1 = ld_frag_b(ldsW0 + (16 + lm) * P1E + kt * 32 + hi * 16);
      d0 = wmma_bf16(a, bf0, d0);
      d1 = wmma_bf16(a, bf1, d1);
    }
    __bf16* hD = h0buf[1];
#pragma unroll
    for (int r = 0; r < 8; ++r) {
      hD[(rbase + r) * HDIM + col0]      = (__bf16)fast_tanh(d0[r]);
      hD[(rbase + r) * HDIM + col0 + 16] = (__bf16)fast_tanh(d1[r]);
    }
  }

#pragma unroll 1
  for (int t = 0; t < T_STEPS; ++t) {
    const int cur = t & 1, nxt = cur ^ 1;
    const int tp1 = (t + 1 < T_STEPS) ? (t + 1) : (T_STEPS - 1);   // clamped

    // prefetch next step's preX fragments while waiting (global_prefetch_b8)
    {
      const __bf16* f = preX + (((long)(tp1 * 4 + wave) * 32 + wg * 2) * 256);
      __builtin_prefetch((const void*)(f + lane * 16), 0, 0);
    }
    grid_barrier(cnt, gen);

    // accumulators: c* = phase2(t) [layer 1], d* = phase1(t+1) [layer 0]
    v8f c0, c1, d0, d1;
#pragma unroll
    for (int r = 0; r < 8; ++r) { c0[r] = bias1_0; c1[r] = bias1_1; }
    {
      const __bf16* f = preX + (((long)(tp1 * 4 + wave) * 32 + wg * 2) * 256 + lane * 8);
      v8bf p0 = *(const v8bf*)f;
      v8bf p1 = *(const v8bf*)(f + 256);
#pragma unroll
      for (int r = 0; r < 8; ++r) { d0[r] = (float)p0[r]; d1[r] = (float)p1[r]; }
    }

    const __bf16* hN = h0buf[nxt] + rowoff;   // new h0 (synced at barrier)
    const __bf16* hO = h1buf[cur] + rowoff;   // old h1
    // K 0..511: A-fragments of h0' feed BOTH layer-1 (c) and next layer-0 (d)
#pragma unroll 4
    for (int kt = 0; kt < 16; ++kt) {
      v16bf a    = ld_frag_a(hN + kt * 32 + hi * 16);
      v16bf w1f0 = ld_frag_b(ldsW1 + lm * P2E        + kt * 32 + hi * 16);
      v16bf w1f1 = ld_frag_b(ldsW1 + (16 + lm) * P2E + kt * 32 + hi * 16);
      v16bf w0f0 = ld_frag_b(ldsW0 + lm * P1E        + kt * 32 + hi * 16);
      v16bf w0f1 = ld_frag_b(ldsW0 + (16 + lm) * P1E + kt * 32 + hi * 16);
      c0 = wmma_bf16(a, w1f0, c0);
      c1 = wmma_bf16(a, w1f1, c1);
      d0 = wmma_bf16(a, w0f0, d0);
      d1 = wmma_bf16(a, w0f1, d1);
    }
    // K 512..1023 of layer 1: A from old h1
#pragma unroll 4
    for (int kt = 16; kt < 32; ++kt) {
      v16bf a    = ld_frag_a(hO + (kt - 16) * 32 + hi * 16);
      v16bf w1f0 = ld_frag_b(ldsW1 + lm * P2E        + kt * 32 + hi * 16);
      v16bf w1f1 = ld_frag_b(ldsW1 + (16 + lm) * P2E + kt * 32 + hi * 16);
      c0 = wmma_bf16(a, w1f0, c0);
      c1 = wmma_bf16(a, w1f1, c1);
    }

    { // epilogue phase2(t): h1' -> h1buf[nxt] (+ final output)
      __bf16* hD = h1buf[nxt];
#pragma unroll
      for (int r = 0; r < 8; ++r) {
        float v0 = fast_tanh(c0[r]);
        float v1 = fast_tanh(c1[r]);
        hD[(rbase + r) * HDIM + col0]      = (__bf16)v0;
        hD[(rbase + r) * HDIM + col0 + 16] = (__bf16)v1;
        if (t == T_STEPS - 1) {
          out[(rbase + r) * HDIM + col0]      = v0;
          out[(rbase + r) * HDIM + col0 + 16] = v1;
        }
      }
    }
    { // epilogue phase1(t+1): h0'' -> h0buf[cur] (dead buffer this interval)
      __bf16* hD = h0buf[cur];
#pragma unroll
      for (int r = 0; r < 8; ++r) {
        hD[(rbase + r) * HDIM + col0]      = (__bf16)fast_tanh(d0[r]);
        hD[(rbase + r) * HDIM + col0 + 16] = (__bf16)fast_tanh(d1[r]);
      }
    }
  }
}

// ---------------------------------------------------------------------------
extern "C" void kernel_launch(void* const* d_in, const int* in_sizes, int n_in,
                              void* d_out, int out_size, void* d_ws, size_t ws_size,
                              hipStream_t stream) {
  const int*   input = (const int*)  d_in[0];
  const float* emb   = (const float*)d_in[1];
  const float* Wih0  = (const float*)d_in[2];
  const float* Whh0  = (const float*)d_in[3];
  const float* bih0  = (const float*)d_in[4];
  const float* bhh0  = (const float*)d_in[5];
  const float* Wih1  = (const float*)d_in[6];
  const float* Whh1  = (const float*)d_in[7];
  const float* bih1  = (const float*)d_in[8];
  const float* bhh1  = (const float*)d_in[9];

  char* w = (char*)d_ws;
  __bf16* preX   = (__bf16*)w; w += (size_t)(T_STEPS * BATCH / 16) * (HDIM / 16) * 512; // 64 MiB
  __bf16* whh0pk = (__bf16*)w; w += (size_t)NWG * NCOL * P1E * 2;                       // 520 KiB
  __bf16* w1pk   = (__bf16*)w; w += (size_t)NWG * NCOL * P2E * 2;                       // 1 MiB
  __bf16* wih0bf = (__bf16*)w; w += (size_t)HDIM * EDIM * 2;                            // 512 KiB
  __bf16* embbf  = (__bf16*)w; w += (size_t)VOCAB * EDIM * 2;                           // 31.25 MiB
  float*  b0     = (float*)w;  w += 2048;
  float*  b1     = (float*)w;  w += 2048;
  __bf16* h0a    = (__bf16*)w; w += (size_t)BATCH * HDIM * 2;
  __bf16* h0b    = (__bf16*)w; w += (size_t)BATCH * HDIM * 2;
  __bf16* h1a    = (__bf16*)w; w += (size_t)BATCH * HDIM * 2;
  __bf16* h1b    = (__bf16*)w; w += (size_t)BATCH * HDIM * 2;
  unsigned* bar  = (unsigned*)w; w += 512;

  // zero h state + barrier (h bufs and bar are contiguous)
  hipMemsetAsync((void*)h0a, 0, (size_t)4 * BATCH * HDIM * 2 + 512, stream);

  pack_kernel<<<4096, 256, 0, stream>>>(emb, Wih0, Whh0, Wih1, Whh1,
                                        bih0, bhh0, bih1, bhh1,
                                        embbf, wih0bf, whh0pk, w1pk, b0, b1);
  prex_kernel<<<dim3(T_STEPS * BATCH / 16, 4), 256, 0, stream>>>(input, embbf, wih0bf, b0, preX);
  rnn_kernel<<<NWG, 128, 0, stream>>>(preX, whh0pk, w1pk, b1,
                                      h0a, h0b, h1a, h1b, bar, (float*)d_out);
}
